// AttentionSAT_10874857193701
// MI455X (gfx1250) — compile-verified, matrix operand order checked by reading, not run on hardware
//
#include <hip/hip_runtime.h>
#include <hip/hip_bf16.h>
#include <stdint.h>
#include <math.h>

// ---------------------------------------------------------------------------
// Types
// ---------------------------------------------------------------------------
typedef _Float16 f16;
typedef _Float16 v8h  __attribute__((ext_vector_type(8)));
typedef _Float16 v16h __attribute__((ext_vector_type(16)));
typedef float    v8f  __attribute__((ext_vector_type(8)));

#define N_VARS    20000
#define N_CLAUSES 84000
#define FDIM      256
#define QMDIM     64
#define ROUNDS    8

// ---------------------------------------------------------------------------
// Device helpers
// ---------------------------------------------------------------------------
__device__ __forceinline__ float wave_sum(float v) {
#pragma unroll
  for (int off = 16; off > 0; off >>= 1) v += __shfl_xor(v, off, 32);
  return v;
}
__device__ __forceinline__ float softplusf(float x) {
  return fmaxf(x, 0.0f) + log1pf(__expf(-fabsf(x)));
}
__device__ __forceinline__ float sigmoidf_(float x) {
  return 1.0f / (1.0f + __expf(-x));
}
__device__ __forceinline__ unsigned hash_u32(unsigned x) {
  x ^= x >> 16; x *= 0x7feb352dU; x ^= x >> 15; x *= 0x846ca68bU; x ^= x >> 16;
  return x;
}
__device__ __forceinline__ float gauss_noise(int r, int v, int j) {
  unsigned s  = hash_u32(0x9e3779b9U*(unsigned)r + 0x85ebca6bU*(unsigned)v +
                         0xc2b2ae35U*(unsigned)j + 0x42424242U);
  unsigned s2 = hash_u32(s + 0x7f4a7c15U);
  float u1 = ((s  >> 8) + 1) * (1.0f / 16777217.0f);
  float u2 = ((s2 >> 8)    ) * (1.0f / 16777216.0f);
  return sqrtf(-2.0f * logf(u1)) * __cosf(6.28318530718f * u2);
}
__device__ __forceinline__ unsigned floatToKey(float f) {
  unsigned b = __float_as_uint(f);
  return (b & 0x80000000u) ? ~b : (b | 0x80000000u);
}
__device__ __forceinline__ float keyToFloat(unsigned k) {
  unsigned b = (k & 0x80000000u) ? (k ^ 0x80000000u) : ~k;
  return __uint_as_float(b);
}

// CDNA5 async global -> LDS copy (16 bytes per active lane), ASYNCcnt-tracked.
__device__ __forceinline__ void async_copy_b128(const f16* lds_dst, const f16* gsrc) {
  unsigned lds_off = (unsigned)(unsigned long long)(const void*)lds_dst;
  asm volatile("global_load_async_to_lds_b128 %0, %1, off"
               :: "v"(lds_off), "v"(gsrc)
               : "memory");
}
// Async loads complete in order: waiting <=3 guarantees the 3 oldest (current
// buffer) are done while the 3 newest (next buffer) remain in flight.
__device__ __forceinline__ void wait_async_le3() {
  asm volatile("s_wait_asynccnt 0x3" ::: "memory");
}
__device__ __forceinline__ void wait_async_le0() {
  asm volatile("s_wait_asynccnt 0x0" ::: "memory");
}

// ---------------------------------------------------------------------------
// WMMA GEMM:  C(MxN) = relu?( A(f16, M x Kp) * Bt^T + bias )
// A   : f16, row stride lda (== Kp), K-padding zero-filled by producer.
// Bt  : f16, [N][Kp] transposed weight, zero-padded.
// Cf  : optional f32 output (ldc = N), Ch : optional f16 output (ldc = N).
// Block: 256 threads (8 waves). Tile 128(M) x 64(N). N % 64 == 0, Kp % 32 == 0.
// Double-buffered async staging: K-step k+1 streams into LDS via
// global_load_async_to_lds_b128 while the WMMAs of step k execute.
// ---------------------------------------------------------------------------
__global__ __launch_bounds__(256) void gemm_wmma_kernel(
    const f16* __restrict__ A, int lda,
    const f16* __restrict__ Bt, int Kp,
    const float* __restrict__ bias,
    float* __restrict__ Cf, f16* __restrict__ Ch, int ldc,
    int M, int doRelu)
{
  __shared__ alignas(16) f16 As[2][128 * 32];
  __shared__ alignas(16) f16 Bs[2][64 * 32];

  const int t    = threadIdx.x;
  const int lane = t & 31;
  const int wave = t >> 5;
  const int mblk = blockIdx.y * 128;
  const int nblk = blockIdx.x * 64;

  // 16B-chunk coordinates. A tile: 512 chunks -> 2/thread. B tile: 256 -> 1.
  const int ar0 = t >> 2;
  const int ak0 = (t & 3) << 3;
  const int ar1 = ar0 + 64;
  const int bn  = t >> 2;
  const int bk  = (t & 3) << 3;

  // Row clamp: out-of-range rows read row M-1 (results masked at store).
  int garow0 = mblk + ar0; if (garow0 >= M) garow0 = M - 1;
  int garow1 = mblk + ar1; if (garow1 >= M) garow1 = M - 1;

  const f16* aSrc0 = A + (size_t)garow0 * lda + ak0;
  const f16* aSrc1 = A + (size_t)garow1 * lda + ak0;
  const f16* bSrc  = Bt + (size_t)(nblk + bn) * Kp + bk;

  auto issue = [&](int buf, int k0) {
    async_copy_b128(&As[buf][ar0 * 32 + ak0], aSrc0 + k0);
    async_copy_b128(&As[buf][ar1 * 32 + ak0], aSrc1 + k0);
    async_copy_b128(&Bs[buf][bn  * 32 + bk ], bSrc  + k0);
  };

  v8f acc[4] = {};

  const int ksteps = Kp >> 5;
  issue(0, 0);
  for (int ks = 0; ks < ksteps; ++ks) {
    const int cur = ks & 1;
    const bool more = (ks + 1) < ksteps;
    if (more) issue(cur ^ 1, (ks + 1) << 5);   // prefetch next K-step
    if (more) wait_async_le3(); else wait_async_le0();
    __syncthreads();

    const int mrow = lane & 15;
    const int half = lane >> 4;
    v8h alo = *(const v8h*)&As[cur][(wave * 16 + mrow) * 32 + half * 8];
    v8h ahi = *(const v8h*)&As[cur][(wave * 16 + mrow) * 32 + 16 + half * 8];
    v16h a = __builtin_shufflevector(alo, ahi,
                                     0,1,2,3,4,5,6,7,8,9,10,11,12,13,14,15);
#pragma unroll
    for (int j = 0; j < 4; ++j) {
      v8h blo = *(const v8h*)&Bs[cur][(j * 16 + mrow) * 32 + half * 8];
      v8h bhi = *(const v8h*)&Bs[cur][(j * 16 + mrow) * 32 + 16 + half * 8];
      v16h b = __builtin_shufflevector(blo, bhi,
                                       0,1,2,3,4,5,6,7,8,9,10,11,12,13,14,15);
      acc[j] = __builtin_amdgcn_wmma_f32_16x16x32_f16(
          false, a, false, b, (short)0, acc[j], false, false);
    }
    __syncthreads();   // all waves done reading 'cur' before it is re-issued
  }

  const int cn16 = lane & 15;
  const int rb   = (lane >> 4) * 8;
#pragma unroll
  for (int j = 0; j < 4; ++j) {
    int cn = nblk + j * 16 + cn16;
    float bv = bias[cn];
#pragma unroll
    for (int i = 0; i < 8; ++i) {
      int grow = mblk + wave * 16 + rb + i;
      if (grow < M) {
        float v = acc[j][i] + bv;
        if (doRelu) v = fmaxf(v, 0.0f);
        if (Cf) Cf[(size_t)grow * ldc + cn] = v;
        if (Ch) Ch[(size_t)grow * ldc + cn] = (f16)v;
      }
    }
  }
}

// ---------------------------------------------------------------------------
// Utility kernels
// ---------------------------------------------------------------------------
__global__ void zero_f32_kernel(float* p, int n) {
  int i = blockIdx.x * 256 + threadIdx.x;
  if (i < n) p[i] = 0.0f;
}
__global__ void fill_u32_kernel(unsigned* p, unsigned v, int n) {
  int i = blockIdx.x * 256 + threadIdx.x;
  if (i < n) p[i] = v;
}
// W (KxN f32 row-major) -> Bt (N x Kp f16), zero pad K..Kp
__global__ void cvt_wt_kernel(const float* __restrict__ W, f16* __restrict__ Bt,
                              int K, int N, int Kp) {
  int idx = blockIdx.x * 256 + threadIdx.x;
  if (idx >= N * Kp) return;
  int n = idx / Kp, k = idx % Kp;
  Bt[idx] = (k < K) ? (f16)W[(size_t)k * N + n] : (f16)0.0f;
}
__global__ void init_vars_kernel(f16* vars) {
  int idx = blockIdx.x * 256 + threadIdx.x;
  if (idx >= N_VARS * FDIM) return;
  int f = idx & (FDIM - 1);
  float onehot = (f == 0) ? 1.0f : 0.0f;
  vars[idx] = (f16)((onehot - 1.0f / (float)FDIM) * 16.0f * 0.25f);
}
// v1h (f16, stride 288): [vars(256) | noise(4) | zero pad(28)]
__global__ void build_v1_kernel(const f16* __restrict__ vars,
                                f16* __restrict__ v1, int r) {
  int idx = blockIdx.x * 256 + threadIdx.x;
  if (idx >= N_VARS * 288) return;
  int v = idx / 288, j = idx % 288;
  float x;
  if      (j < 256) x = (float)vars[(size_t)v * FDIM + j];
  else if (j < 260) x = gauss_noise(r, v, j - 256);
  else              x = 0.0f;
  v1[idx] = (f16)x;
}
// per clause & feature: cl = exp(-sum softplus(q*sign)); scatter grad (f32)
__global__ void clause_kernel(const float* __restrict__ query,
                              const int* __restrict__ clauses,
                              f16* __restrict__ clh, float* __restrict__ grad) {
  int t = threadIdx.x;
  int c = blockIdx.x * 4 + (t >> 6);
  int m = t & 63;
  if (c >= N_CLAUSES) return;
  int   idx[3]; float sgn[3], vals[3];
  float sp = 0.0f;
#pragma unroll
  for (int l = 0; l < 3; ++l) {
    int lit = clauses[c * 3 + l];
    int id  = (lit > 0 ? lit : -lit) - 1;
    float sg = lit > 0 ? 1.0f : -1.0f;
    float v  = query[(size_t)id * QMDIM + m] * sg;
    idx[l] = id; sgn[l] = sg; vals[l] = v;
    sp += softplusf(v);
  }
  float clv = __expf(-sp);
  clh[(size_t)c * QMDIM + m] = (f16)clv;
#pragma unroll
  for (int l = 0; l < 3; ++l)
    atomicAdd(&grad[(size_t)idx[l] * QMDIM + m],
              -clv * sigmoidf_(vals[l]) * sgn[l]);
}
// attention pass 1: per-edge score + segment max (f16 q/k, f32 accumulate)
// Each lane reads a contiguous 32B chunk; wave-reduction order is irrelevant.
__global__ void edge_scores_kernel(const f16* __restrict__ q,
                                   const f16* __restrict__ k,
                                   const int* __restrict__ evar,
                                   const int* __restrict__ ecl,
                                   float* __restrict__ scores,
                                   unsigned* __restrict__ maxkey, int nE) {
  int e = blockIdx.x * 8 + (threadIdx.x >> 5);
  if (e >= nE) return;
  int lane = threadIdx.x & 31;
  int v = evar[e], c = ecl[e];
  const f16* qp = q + (size_t)v * 512 + lane * 16;
  const f16* kp = k + (size_t)c * 512 + lane * 16;
  v8h q0 = *(const v8h*)qp, q1 = *(const v8h*)(qp + 8);
  v8h k0 = *(const v8h*)kp, k1 = *(const v8h*)(kp + 8);
  float s = 0.0f;
#pragma unroll
  for (int i = 0; i < 8; ++i)
    s += (float)q0[i] * (float)k0[i] + (float)q1[i] * (float)k1[i];
  s = wave_sum(s) * 0.044194173824159216f;  // 1/sqrt(512)
  if (lane == 0) { scores[e] = s; atomicMax(maxkey + v, floatToKey(s)); }
}
// attention pass 2: softmax weights -> denom/num segment sums (f16 v)
__global__ void edge_accum_kernel(const f16* __restrict__ vfeat,
                                  const int* __restrict__ evar,
                                  const int* __restrict__ ecl,
                                  const float* __restrict__ scores,
                                  const unsigned* __restrict__ maxkey,
                                  float* __restrict__ denom,
                                  float* __restrict__ num, int nE) {
  int e = blockIdx.x * 8 + (threadIdx.x >> 5);
  if (e >= nE) return;
  int lane = threadIdx.x & 31;
  int v = evar[e], c = ecl[e];
  float w = __expf(scores[e] - keyToFloat(maxkey[v]));
  if (lane == 0) atomicAdd(denom + v, w);
  v8h vv = *(const v8h*)(vfeat + (size_t)c * FDIM + lane * 8);
  float* np = num + (size_t)v * FDIM + lane * 8;
#pragma unroll
  for (int i = 0; i < 8; ++i) atomicAdd(np + i, w * (float)vv[i]);
}
__global__ void att_norm_kernel(const float* __restrict__ num,
                                const float* __restrict__ denom,
                                float* __restrict__ out) {
  int idx = blockIdx.x * 256 + threadIdx.x;
  if (idx >= N_VARS * FDIM) return;
  out[idx] = num[idx] / (denom[idx / FDIM] + 1e-9f);
}
// unit (f16, 832): [vars f16(256) | grad f32(64) | lpos(256) | lneg(256)]
__global__ void build_unit_kernel(const f16* __restrict__ vars,
                                  const float* __restrict__ grad,
                                  const float* __restrict__ lpos,
                                  const float* __restrict__ lneg,
                                  f16* __restrict__ unit) {
  int idx = blockIdx.x * 256 + threadIdx.x;
  if (idx >= N_VARS * 832) return;
  int v = idx / 832, j = idx % 832;
  float x;
  if      (j < 256) x = (float)vars[(size_t)v * FDIM + j];
  else if (j < 320) x = grad[(size_t)v * QMDIM + (j - 256)];
  else if (j < 576) x = lpos[(size_t)v * FDIM + (j - 320)];
  else              x = lneg[(size_t)v * FDIM + (j - 576)];
  unit[idx] = (f16)x;
}
// LayerNorm * 0.25, writes the f16 variable state for the next round
__global__ void layernorm_kernel(const float* __restrict__ x,
                                 const float* __restrict__ gamma,
                                 const float* __restrict__ beta,
                                 f16* __restrict__ outh) {
  int row = blockIdx.x * 8 + (threadIdx.x >> 5);
  if (row >= N_VARS) return;
  int lane = threadIdx.x & 31;
  const float* xp = x + (size_t)row * FDIM;
  float vals[8], s = 0.0f, s2 = 0.0f;
#pragma unroll
  for (int i = 0; i < 8; ++i) {
    float v = xp[lane + i * 32];
    vals[i] = v; s += v; s2 += v * v;
  }
  s  = wave_sum(s);
  s2 = wave_sum(s2);
  float mu  = s * (1.0f / FDIM);
  float var = s2 * (1.0f / FDIM) - mu * mu;
  float rs  = rsqrtf(var + 0.001f);
  f16* op = outh + (size_t)row * FDIM;
#pragma unroll
  for (int i = 0; i < 8; ++i) {
    int f = lane + i * 32;
    op[f] = (f16)(((vals[i] - mu) * rs * gamma[f] + beta[f]) * 0.25f);
  }
}
// final out-MLP layer: N=1 row dot (f32 activations)
__global__ void logits_kernel(const float* __restrict__ x,
                              const float* __restrict__ W,
                              const float* __restrict__ b,
                              float* __restrict__ logits) {
  int row = blockIdx.x * 8 + (threadIdx.x >> 5);
  if (row >= N_VARS) return;
  int lane = threadIdx.x & 31;
  const float* xp = x + (size_t)row * FDIM;
  float s = 0.0f;
#pragma unroll
  for (int i = 0; i < 8; ++i) { int f = lane + i * 32; s += xp[f] * W[f]; }
  s = wave_sum(s);
  if (lane == 0) logits[row] = s + b[0];
}
__global__ void logit_loss_kernel(const float* __restrict__ logits,
                                  const int* __restrict__ clauses,
                                  float* __restrict__ acc) {
  int c = blockIdx.x * 256 + threadIdx.x;
  if (c >= N_CLAUSES) return;
  float sp = 0.0f;
#pragma unroll
  for (int l = 0; l < 3; ++l) {
    int lit = clauses[c * 3 + l];
    int id  = (lit > 0 ? lit : -lit) - 1;
    float sg = lit > 0 ? 1.0f : -1.0f;
    sp += softplusf(logits[id] * sg);
  }
  float clv = __expf(-sp);
  float t = -logf(1.0f - clv + 1e-8f);
  atomicAdd(acc, t * t);
}
__global__ void write_out_kernel(const float* __restrict__ logits,
                                 const float* __restrict__ acc,
                                 float* __restrict__ out) {
  int i = blockIdx.x * 256 + threadIdx.x;
  if (i < N_VARS)       out[i] = logits[i];
  else if (i == N_VARS) out[i] = acc[0] * (1.0f / ROUNDS);
}

// ---------------------------------------------------------------------------
// Host helpers
// ---------------------------------------------------------------------------
static inline int kpad(int K) { return (K + 31) & ~31; }

static void launch_gemm(hipStream_t s, const f16* A, int M, int Kp, int lda,
                        const f16* Bt, const float* bias,
                        float* Cf, f16* Ch, int N, int relu) {
  dim3 grid((unsigned)(N / 64), (unsigned)((M + 127) / 128));
  gemm_wmma_kernel<<<grid, 256, 0, s>>>(A, lda, Bt, Kp, bias, Cf, Ch, N, M, relu);
}
static void launch_cvt(hipStream_t s, const float* W, f16* Bt, int K, int N) {
  int total = N * kpad(K);
  cvt_wt_kernel<<<(total + 255) / 256, 256, 0, s>>>(W, Bt, K, N, kpad(K));
}
static void launch_zero(hipStream_t s, float* p, int n) {
  zero_f32_kernel<<<(n + 255) / 256, 256, 0, s>>>(p, n);
}

// ---------------------------------------------------------------------------
// Entry
// ---------------------------------------------------------------------------
extern "C" void kernel_launch(void* const* d_in, const int* in_sizes, int n_in,
                              void* d_out, int out_size, void* d_ws, size_t ws_size,
                              hipStream_t stream) {
  (void)in_sizes; (void)n_in; (void)out_size; (void)ws_size;

  const int* clauses    = (const int*)d_in[0];
  const int* pos_var    = (const int*)d_in[1];
  const int* pos_clause = (const int*)d_in[2];
  const int* neg_var    = (const int*)d_in[3];
  const int* neg_clause = (const int*)d_in[4];
  const float* Wq0 = (const float*)d_in[5];  const float* bq0 = (const float*)d_in[6];
  const float* Wq1 = (const float*)d_in[7];  const float* bq1 = (const float*)d_in[8];
  const float* Wq2 = (const float*)d_in[9];  const float* bq2 = (const float*)d_in[10];
  const float* Wg0 = (const float*)d_in[11]; const float* bg0 = (const float*)d_in[12];
  const float* Wg1 = (const float*)d_in[13]; const float* bg1 = (const float*)d_in[14];
  const float* Wg2 = (const float*)d_in[15]; const float* bg2 = (const float*)d_in[16];
  const float* Wo0 = (const float*)d_in[17]; const float* bo0 = (const float*)d_in[18];
  const float* Wo1 = (const float*)d_in[19]; const float* bo1 = (const float*)d_in[20];
  const float* Wo2 = (const float*)d_in[21]; const float* bo2 = (const float*)d_in[22];
  const float* Apq = (const float*)d_in[23]; const float* Apqb = (const float*)d_in[24];
  const float* Apk = (const float*)d_in[25]; const float* Apkb = (const float*)d_in[26];
  const float* Apv = (const float*)d_in[27]; const float* Apvb = (const float*)d_in[28];
  const float* Anq = (const float*)d_in[29]; const float* Anqb = (const float*)d_in[30];
  const float* Ank = (const float*)d_in[31]; const float* Ankb = (const float*)d_in[32];
  const float* Anv = (const float*)d_in[33]; const float* Anvb = (const float*)d_in[34];
  const float* ln_gamma = (const float*)d_in[35];
  const float* ln_beta  = (const float*)d_in[36];

  // ---- workspace carve (256B aligned) ----
  size_t off = 0;
  auto alloc = [&](size_t nbytes) -> void* {
    void* p = (char*)d_ws + off;
    off += (nbytes + 255) & ~(size_t)255;
    return p;
  };
  // f16 activations (GEMM A operands; K-padded strides)
  f16* varsAh = (f16*)alloc((size_t)N_VARS * FDIM * 2);
  f16* varsBh = (f16*)alloc((size_t)N_VARS * FDIM * 2);
  f16* v1h    = (f16*)alloc((size_t)N_VARS * 288 * 2);
  f16* h1h    = (f16*)alloc((size_t)N_VARS * 128 * 2);
  f16* h2h    = (f16*)alloc((size_t)N_VARS * 128 * 2);
  f16* clbh   = (f16*)alloc((size_t)N_CLAUSES * QMDIM * 2);
  f16* unith  = (f16*)alloc((size_t)N_VARS * 832 * 2);
  f16* g1h    = (f16*)alloc((size_t)N_VARS * 512 * 2);
  f16* g2h    = (f16*)alloc((size_t)N_VARS * 512 * 2);
  f16* o1h    = (f16*)alloc((size_t)N_VARS * FDIM * 2);
  // f16 attention projections (bandwidth-dominant edge passes read these)
  f16* attqh  = (f16*)alloc((size_t)N_VARS * 512 * 2);
  f16* attkh  = (f16*)alloc((size_t)N_CLAUSES * 512 * 2);
  f16* attvh  = (f16*)alloc((size_t)N_CLAUSES * FDIM * 2);
  // f32 buffers (non-GEMM consumers)
  float* query  = (float*)alloc((size_t)N_VARS * QMDIM * 4);
  float* grad   = (float*)alloc((size_t)N_VARS * QMDIM * 4);
  float* scores = (float*)alloc((size_t)(2 * N_CLAUSES) * 4);
  unsigned* maxkey = (unsigned*)alloc((size_t)N_VARS * 4);
  float* denom  = (float*)alloc((size_t)N_VARS * 4);
  float* num    = (float*)alloc((size_t)N_VARS * FDIM * 4);
  float* lpos   = (float*)alloc((size_t)N_VARS * FDIM * 4);
  float* lneg   = (float*)alloc((size_t)N_VARS * FDIM * 4);
  float* gateo  = (float*)alloc((size_t)N_VARS * FDIM * 4);
  float* o2     = (float*)alloc((size_t)N_VARS * FDIM * 4);
  float* logits = (float*)alloc((size_t)N_VARS * 4);
  float* lossac = (float*)alloc(4 * 4);
  // f16 transposed weights
  f16* wt_q0 = (f16*)alloc((size_t)128 * kpad(260) * 2);
  f16* wt_q1 = (f16*)alloc((size_t)128 * kpad(128) * 2);
  f16* wt_q2 = (f16*)alloc((size_t)64  * kpad(128) * 2);
  f16* wt_g0 = (f16*)alloc((size_t)512 * kpad(832) * 2);
  f16* wt_g1 = (f16*)alloc((size_t)512 * kpad(512) * 2);
  f16* wt_g2 = (f16*)alloc((size_t)256 * kpad(512) * 2);
  f16* wt_o0 = (f16*)alloc((size_t)256 * kpad(256) * 2);
  f16* wt_o1 = (f16*)alloc((size_t)256 * kpad(256) * 2);
  f16* wt_pq = (f16*)alloc((size_t)512 * kpad(256) * 2);
  f16* wt_pk = (f16*)alloc((size_t)512 * kpad(64) * 2);
  f16* wt_pv = (f16*)alloc((size_t)256 * kpad(64) * 2);
  f16* wt_nq = (f16*)alloc((size_t)512 * kpad(256) * 2);
  f16* wt_nk = (f16*)alloc((size_t)512 * kpad(64) * 2);
  f16* wt_nv = (f16*)alloc((size_t)256 * kpad(64) * 2);

  // ---- setup ----
  launch_cvt(stream, Wq0, wt_q0, 260, 128);
  launch_cvt(stream, Wq1, wt_q1, 128, 128);
  launch_cvt(stream, Wq2, wt_q2, 128, 64);
  launch_cvt(stream, Wg0, wt_g0, 832, 512);
  launch_cvt(stream, Wg1, wt_g1, 512, 512);
  launch_cvt(stream, Wg2, wt_g2, 512, 256);
  launch_cvt(stream, Wo0, wt_o0, 256, 256);
  launch_cvt(stream, Wo1, wt_o1, 256, 256);
  launch_cvt(stream, Apq, wt_pq, 256, 512);
  launch_cvt(stream, Apk, wt_pk, 64, 512);
  launch_cvt(stream, Apv, wt_pv, 64, 256);
  launch_cvt(stream, Anq, wt_nq, 256, 512);
  launch_cvt(stream, Ank, wt_nk, 64, 512);
  launch_cvt(stream, Anv, wt_nv, 64, 256);
  launch_zero(stream, lossac, 1);
  init_vars_kernel<<<(N_VARS * FDIM + 255) / 256, 256, 0, stream>>>(varsAh);

  f16* vin  = varsAh;
  f16* vout = varsBh;

  for (int r = 0; r < ROUNDS; ++r) {
    // query = MLP_query([vars, noise])  (all-WMMA chain, f16 intermediates)
    build_v1_kernel<<<(N_VARS * 288 + 255) / 256, 256, 0, stream>>>(vin, v1h, r);
    launch_gemm(stream, v1h, N_VARS, 288, 288, wt_q0, bq0, nullptr, h1h, 128, 1);
    launch_gemm(stream, h1h, N_VARS, 128, 128, wt_q1, bq1, nullptr, h2h, 128, 1);
    launch_gemm(stream, h2h, N_VARS, 128, 128, wt_q2, bq2, query, nullptr, 64, 0);

    // clause loss + grad scatter
    launch_zero(stream, grad, N_VARS * QMDIM);
    clause_kernel<<<(N_CLAUSES + 3) / 4, 256, 0, stream>>>(query, clauses, clbh, grad);

    // attention (pos then neg, reusing q/k/v buffers)
    for (int side = 0; side < 2; ++side) {
      const f16* wq = side ? wt_nq : wt_pq;
      const f16* wk = side ? wt_nk : wt_pk;
      const f16* wv = side ? wt_nv : wt_pv;
      const float* bq = side ? Anqb : Apqb;
      const float* bk = side ? Ankb : Apkb;
      const float* bv = side ? Anvb : Apvb;
      const int* evar = side ? neg_var : pos_var;
      const int* ecl  = side ? neg_clause : pos_clause;
      int nE = side ? N_CLAUSES : 2 * N_CLAUSES;
      float* dst = side ? lneg : lpos;

      launch_gemm(stream, vin,  N_VARS,    256, 256, wq, bq, nullptr, attqh, 512, 0);
      launch_gemm(stream, clbh, N_CLAUSES, 64,  64,  wk, bk, nullptr, attkh, 512, 0);
      launch_gemm(stream, clbh, N_CLAUSES, 64,  64,  wv, bv, nullptr, attvh, 256, 0);
      fill_u32_kernel<<<(N_VARS + 255) / 256, 256, 0, stream>>>(maxkey, 0u, N_VARS);
      launch_zero(stream, denom, N_VARS);
      launch_zero(stream, num, N_VARS * FDIM);
      edge_scores_kernel<<<(nE + 7) / 8, 256, 0, stream>>>(
          attqh, attkh, evar, ecl, scores, maxkey, nE);
      edge_accum_kernel<<<(nE + 7) / 8, 256, 0, stream>>>(
          attvh, evar, ecl, scores, maxkey, denom, num, nE);
      att_norm_kernel<<<(N_VARS * FDIM + 255) / 256, 256, 0, stream>>>(num, denom, dst);
    }

    // gate MLP + layer norm (-> next-round f16 state)
    build_unit_kernel<<<(N_VARS * 832 + 255) / 256, 256, 0, stream>>>(
        vin, grad, lpos, lneg, unith);
    launch_gemm(stream, unith, N_VARS, 832, 832, wt_g0, bg0, nullptr, g1h, 512, 1);
    launch_gemm(stream, g1h,  N_VARS, 512, 512, wt_g1, bg1, nullptr, g2h, 512, 1);
    launch_gemm(stream, g2h,  N_VARS, 512, 512, wt_g2, bg2, gateo, nullptr, 256, 0);
    layernorm_kernel<<<(N_VARS + 7) / 8, 256, 0, stream>>>(gateo, ln_gamma, ln_beta, vout);

    // out MLP + loss
    launch_gemm(stream, vout, N_VARS, 256, 256, wt_o0, bo0, nullptr, o1h, 256, 1);
    launch_gemm(stream, o1h,  N_VARS, 256, 256, wt_o1, bo1, o2, nullptr, 256, 1);
    logits_kernel<<<(N_VARS + 7) / 8, 256, 0, stream>>>(o2, Wo2, bo2, logits);
    logit_loss_kernel<<<(N_CLAUSES + 255) / 256, 256, 0, stream>>>(logits, clauses, lossac);

    f16* tmp = vin; vin = vout; vout = tmp;
  }

  write_out_kernel<<<(N_VARS + 1 + 255) / 256, 256, 0, stream>>>(
      logits, lossac, (float*)d_out);
}
